// GCN_38946763440877
// MI455X (gfx1250) — matrix operand order repaired and checked by reading, not compiled, not run on hardware
//
#include <hip/hip_runtime.h>
#include <hip/hip_bf16.h>
#include <cstdint>

#define D 768
typedef __attribute__((ext_vector_type(16))) __bf16   v16bf;
typedef __attribute__((ext_vector_type(8)))  float    v8f;
typedef __attribute__((ext_vector_type(16))) unsigned short v16u;
typedef __attribute__((ext_vector_type(8)))  unsigned short v8u;

// Round-to-nearest-even split of fp32 into hi/lo bf16 (a ~= hi + lo).
__device__ __forceinline__ void bf16_split(float x, unsigned short& hi, unsigned short& lo) {
    unsigned u = __float_as_uint(x);
    unsigned r = u + 0x7FFFu + ((u >> 16) & 1u);
    hi = (unsigned short)(r >> 16);
    float fhi = __uint_as_float(((unsigned)hi) << 16);
    float res = x - fhi;
    unsigned u2 = __float_as_uint(res);
    unsigned r2 = u2 + 0x7FFFu + ((u2 >> 16) & 1u);
    lo = (unsigned short)(r2 >> 16);
}

// ---------------- degree / norm ----------------
__global__ void k_deg_init(unsigned* deg, int n) {
    int i = blockIdx.x * blockDim.x + threadIdx.x;
    if (i < n) deg[i] = 1u;                       // self-loop
}
__global__ void k_deg_count(const int* __restrict__ dst, unsigned* deg, int e) {
    int i = blockIdx.x * blockDim.x + threadIdx.x;
    if (i < e) atomicAdd(&deg[dst[i]], 1u);
}
__global__ void k_dinv(const unsigned* __restrict__ deg, float* __restrict__ dinv, int n) {
    int i = blockIdx.x * blockDim.x + threadIdx.x;
    if (i < n) dinv[i] = rsqrtf((float)deg[i]);
}

// ---------------- weight transpose + bf16 split: Wt[n][k] = W[k][n] ----------------
__global__ void k_prep_w(const float* __restrict__ W,
                         unsigned short* __restrict__ wt_hi,
                         unsigned short* __restrict__ wt_lo) {
    int idx = blockIdx.x * blockDim.x + threadIdx.x;
    if (idx >= D * D) return;
    int n = idx / D, k = idx - n * D;
    unsigned short h, l;
    bf16_split(W[(size_t)k * D + n], h, l);
    wt_hi[idx] = h;
    wt_lo[idx] = l;
}

// ---------------- activation bf16 split (row-major, k contiguous) ----------------
__global__ void k_prep_x(const float* __restrict__ x,
                         unsigned short* __restrict__ xh,
                         unsigned short* __restrict__ xl, size_t total) {
    size_t i = (size_t)blockIdx.x * blockDim.x + threadIdx.x;
    if (i >= total) return;
    unsigned short h, l;
    bf16_split(x[i], h, l);
    xh[i] = h;
    xl[i] = l;
}

// ---------------- GEMM: out[m][n] = dinv[m] * sum_k A[m][k] * W[k][n] ----------------
// A pre-split to bf16 hi/lo; one wave computes a 16x16 tile with
// v_wmma_f32_16x16x32_bf16 (split-bf16: hi*hi + lo*hi + hi*lo, fp32 accumulate).
__global__ __launch_bounds__(256) void k_gemm_scale(
    const unsigned short* __restrict__ a_hi,    // [M, D] row-major
    const unsigned short* __restrict__ a_lo,    // [M, D] row-major
    const unsigned short* __restrict__ wt_hi,   // [D, D] n-major
    const unsigned short* __restrict__ wt_lo,   // [D, D] n-major
    const float* __restrict__ dinv,             // [M]
    float* __restrict__ out,                    // [M, D]
    int M)
{
    const int lane = threadIdx.x & 31;
    const int wave = threadIdx.x >> 5;
    const int m0 = blockIdx.x * 16;
    const int n0 = (blockIdx.y * 8 + wave) * 16;
    const int half = lane >> 4;          // 0: lanes 0-15, 1: lanes 16-31
    const int l15  = lane & 15;

    // A contiguous layout (matches B): elems 0..15 -> K=k+aoff+0..15
    const size_t arow = (size_t)(m0 + l15) * D;
    const int aoff = half * 16;
    // B contiguous layout (matches A): elems 0..15 -> K=k+boff+0..15, col n0+l15
    const int bcol = n0 + l15;
    const int boff = half * 16;
    const unsigned short* ah_p = a_hi  + arow + aoff;
    const unsigned short* al_p = a_lo  + arow + aoff;
    const unsigned short* bh_p = wt_hi + (size_t)bcol * D + boff;
    const unsigned short* bl_p = wt_lo + (size_t)bcol * D + boff;

    v8f acc = {};
    for (int k = 0; k < D; k += 32) {
        v8u ah0 = *(const v8u*)(ah_p + k);
        v8u ah1 = *(const v8u*)(ah_p + k + 8);
        v8u al0 = *(const v8u*)(al_p + k);
        v8u al1 = *(const v8u*)(al_p + k + 8);
        v16bf ah = __builtin_bit_cast(v16bf, __builtin_shufflevector(ah0, ah1,
                       0,1,2,3,4,5,6,7,8,9,10,11,12,13,14,15));
        v16bf al = __builtin_bit_cast(v16bf, __builtin_shufflevector(al0, al1,
                       0,1,2,3,4,5,6,7,8,9,10,11,12,13,14,15));
        v16bf bh = __builtin_bit_cast(v16bf, *(const v16u*)(bh_p + k));
        v16bf bl = __builtin_bit_cast(v16bf, *(const v16u*)(bl_p + k));

        acc = __builtin_amdgcn_wmma_f32_16x16x32_bf16(false, ah, false, bh, (short)0, acc, false, false);
        acc = __builtin_amdgcn_wmma_f32_16x16x32_bf16(false, al, false, bh, (short)0, acc, false, false);
        acc = __builtin_amdgcn_wmma_f32_16x16x32_bf16(false, ah, false, bl, (short)0, acc, false, false);
    }

    // C/D layout: VGPR r -> row m0 + half*8 + r, col n0 + l15
    const int orow0 = m0 + half * 8;
    const int ocol  = n0 + l15;
    if (m0 + 16 <= M) {                      // uniform fast path (M % 16 == 0 case)
#pragma unroll
        for (int r = 0; r < 8; ++r) {
            int row = orow0 + r;
            out[(size_t)row * D + ocol] = acc[r] * dinv[row];
        }
    } else {
#pragma unroll
        for (int r = 0; r < 8; ++r) {
            int row = orow0 + r;
            if (row < M) out[(size_t)row * D + ocol] = acc[r] * dinv[row];
        }
    }
}

// ---------------- edge scatter-add: acc[dst] += hs[src] (self-loops appended) ----------------
__global__ __launch_bounds__(256) void k_scatter(
    const float* __restrict__ hs, const int* __restrict__ src, const int* __restrict__ dst,
    int nedges, int nnodes, float* __restrict__ acc)
{
    int wid  = blockIdx.x * (blockDim.x >> 5) + (threadIdx.x >> 5);
    int lane = threadIdx.x & 31;
    int total = nedges + nnodes;
    if (wid >= total) return;
    int s, d;
    if (wid < nedges) { s = src[wid]; d = dst[wid]; }
    else              { s = d = wid - nedges; }
    const float* hrow = hs  + (size_t)s * D;
    float*       arow = acc + (size_t)d * D;
#pragma unroll
    for (int i = 0; i < 6; ++i) {
        int off = i * 128 + lane * 4;
        float4 v = *(const float4*)(hrow + off);
        unsafeAtomicAdd(arow + off + 0, v.x);
        unsafeAtomicAdd(arow + off + 1, v.y);
        unsafeAtomicAdd(arow + off + 2, v.z);
        unsafeAtomicAdd(arow + off + 3, v.w);
    }
}

// ---- layer-1 epilogue: v = relu(acc*dinv + b); write split bf16 (next layer's A) ----
__global__ void k_final_split(const float* __restrict__ acc, const float* __restrict__ dinv,
                              const float* __restrict__ b,
                              unsigned short* __restrict__ outh,
                              unsigned short* __restrict__ outl, int n)
{
    size_t i = (size_t)blockIdx.x * blockDim.x + threadIdx.x;
    if (i >= (size_t)n * D) return;
    int row = (int)(i / D);
    int col = (int)(i - (size_t)row * D);
    float v = fmaxf(acc[i] * dinv[row] + b[col], 0.0f);
    unsigned short h, l;
    bf16_split(v, h, l);
    outh[i] = h;
    outl[i] = l;
}

// ---- layer-2 epilogue: out = acc*dinv + b (fp32) ----
__global__ void k_final_f32(const float* __restrict__ acc, const float* __restrict__ dinv,
                            const float* __restrict__ b, float* __restrict__ out, int n)
{
    size_t i = (size_t)blockIdx.x * blockDim.x + threadIdx.x;
    if (i >= (size_t)n * D) return;
    int row = (int)(i / D);
    int col = (int)(i - (size_t)row * D);
    out[i] = acc[i] * dinv[row] + b[col];
}

extern "C" void kernel_launch(void* const* d_in, const int* in_sizes, int n_in,
                              void* d_out, int out_size, void* d_ws, size_t ws_size,
                              hipStream_t stream) {
    const float* x  = (const float*)d_in[0];
    const float* W1 = (const float*)d_in[1];
    const float* b1 = (const float*)d_in[2];
    const float* W2 = (const float*)d_in[3];
    const float* b2 = (const float*)d_in[4];
    const int*   ei = (const int*)d_in[5];

    const int N  = in_sizes[0] / D;      // 10000
    const int NE = in_sizes[5] / 2;      // 160000
    const int* src = ei;
    const int* dst = ei + NE;

    // workspace carve-up (256B aligned)
    char* ws = (char*)d_ws;
    size_t off = 0;
    auto carve = [&](size_t bytes) -> char* {
        char* p = ws + off;
        off = (off + bytes + 255) & ~(size_t)255;
        return p;
    };
    unsigned*       deg  = (unsigned*)      carve((size_t)N * 4);
    float*          dinv = (float*)         carve((size_t)N * 4);
    unsigned short* w1h  = (unsigned short*)carve((size_t)D * D * 2);
    unsigned short* w1l  = (unsigned short*)carve((size_t)D * D * 2);
    unsigned short* w2h  = (unsigned short*)carve((size_t)D * D * 2);
    unsigned short* w2l  = (unsigned short*)carve((size_t)D * D * 2);
    unsigned short* xh   = (unsigned short*)carve((size_t)N * D * 2);  // split activations
    unsigned short* xl   = (unsigned short*)carve((size_t)N * D * 2);  // (reused between layers)
    float*          hs   = (float*)         carve((size_t)N * D * 4);  // gemm out (reused)
    float*          acc  = (float*)         carve((size_t)N * D * 4);  // scatter accumulator

    const size_t ND = (size_t)N * D;
    const int fgrid = (int)((ND + 255) / 256);

    // normalization
    k_deg_init <<<(N + 255) / 256, 256, 0, stream>>>(deg, N);
    k_deg_count<<<(NE + 255) / 256, 256, 0, stream>>>(dst, deg, NE);
    k_dinv     <<<(N + 255) / 256, 256, 0, stream>>>(deg, dinv, N);

    // weight transpose + bf16 split; input activation split
    k_prep_w<<<(D * D + 255) / 256, 256, 0, stream>>>(W1, w1h, w1l);
    k_prep_w<<<(D * D + 255) / 256, 256, 0, stream>>>(W2, w2h, w2l);
    k_prep_x<<<fgrid, 256, 0, stream>>>(x, xh, xl, ND);

    dim3 ggrid((N + 15) / 16, D / 16 / 8);   // 625 x 6, 8 waves/block
    int totalw = NE + N;
    int sgrid  = (totalw + 7) / 8;

    // ---- layer 1 ----
    k_gemm_scale<<<ggrid, 256, 0, stream>>>(xh, xl, w1h, w1l, dinv, hs, N);
    hipMemsetAsync(acc, 0, ND * 4, stream);
    k_scatter<<<sgrid, 256, 0, stream>>>(hs, src, dst, NE, N, acc);
    k_final_split<<<fgrid, 256, 0, stream>>>(acc, dinv, b1, xh, xl, N);   // relu -> split x2

    // ---- layer 2 ----
    k_gemm_scale<<<ggrid, 256, 0, stream>>>(xh, xl, w2h, w2l, dinv, hs, N);
    hipMemsetAsync(acc, 0, ND * 4, stream);
    k_scatter<<<sgrid, 256, 0, stream>>>(hs, src, dst, NE, N, acc);
    k_final_f32<<<fgrid, 256, 0, stream>>>(acc, dinv, b2, (float*)d_out, N);
}
